// GRU_SLIMX_9320079033021
// MI455X (gfx1250) — compile-verified, matrix operand order checked by reading, not run on hardware
//
#include <hip/hip_runtime.h>

#define BDIM 16384
#define INDIM 512
#define HDIM 1024
#define LDS_STRIDE 40  // bf16 elems per LDS row: 32 data + 8 pad (80B -> 16B-aligned chunks)

typedef __attribute__((ext_vector_type(16))) __bf16 v16bf;
typedef __attribute__((ext_vector_type(8)))  float  v8f;
typedef __attribute__((ext_vector_type(4)))  unsigned int v4u;
typedef __attribute__((ext_vector_type(8)))  int    v8i;
typedef __attribute__((ext_vector_type(4)))  int    v4i;
typedef __attribute__((ext_vector_type(4)))  float  v4f;
typedef __attribute__((ext_vector_type(4)))  unsigned short v4us;

__device__ __forceinline__ unsigned short f2bf(float f) {
    unsigned int u = __float_as_uint(f);
    u += 0x7FFFu + ((u >> 16) & 1u);   // round-to-nearest-even
    return (unsigned short)(u >> 16);
}

__device__ __forceinline__ float sigmoidf(float x) {
    return 1.0f / (1.0f + __expf(-x));
}

union Frag { v16bf v; v4u q[2]; };

// Load one 16x32 bf16 WMMA operand fragment (A or B^T layout) from an LDS tile.
// Lane l: row = base + (l&15); K-chunks at c0 and c0+16 where c0 = (l>=16 ? 8 : 0).
__device__ __forceinline__ v16bf load_frag(const unsigned short* s, int row, int c0) {
    Frag f;
    const unsigned short* p = s + row * LDS_STRIDE + c0;
    f.q[0] = *(const v4u*)(p);
    f.q[1] = *(const v4u*)(p + 16);
    return f.v;
}

// LDS byte offset of a __shared__ object: the LDS aperture keeps the byte
// offset in addr[31:0] (ISA 10.2), so truncating the flat pointer works.
__device__ __forceinline__ unsigned lds_off_of(const void* p) {
    return (unsigned)(uintptr_t)p;
}

// ---------------------------------------------------------------------------
// Tensor Data Mover: async global->LDS copy of a (rows x 32) bf16 tile whose
// rows are ld elements apart in memory, depositing LDS rows padded to
// LDS_STRIDE (pad_interval = 16 DWORDs = 64B row, pad_amount = 4 DWORDs).
// D# bitfields per CDNA5 ISA 8.3/8.4; groups 2/3 unused (2D tensor).
// ---------------------------------------------------------------------------
__device__ __forceinline__ void tdm_load_tile(const unsigned short* __restrict__ gsrc,
                                              unsigned lds_byte_off, int rows, int ld) {
    unsigned long long ga = (unsigned long long)(uintptr_t)gsrc;
    v4u g0;
    g0.x = 1u;                                               // count=1 (valid user D#)
    g0.y = lds_byte_off;                                     // lds_addr
    g0.z = (unsigned)ga;                                     // global_addr[31:0]
    g0.w = (unsigned)((ga >> 32) & 0x01ffffffull)            // global_addr[56:32]
         | (2u << 30);                                       // type = 2 ("image")
    unsigned uld = (unsigned)ld, urw = (unsigned)rows;
    v8i g1 = {
        (int)((1u << 16)                                     // data_size = 2 bytes
            | (1u << 20)                                     // pad_enable
            | (3u << 22)                                     // pad_interval: 16 DWORDs
            | (3u << 25)),                                   // pad_amount:   4 DWORDs
        (int)((uld & 0xffffu) << 16),                        // tensor_dim0[15:0]
        (int)(((uld >> 16) & 0xffffu) | ((urw & 0xffffu) << 16)),  // dim0[31:16] | dim1[15:0]
        (int)(((urw >> 16) & 0xffffu) | (32u << 16)),        // dim1[31:16] | tile_dim0=32
        (int)(urw & 0xffffu),                                // tile_dim1 = rows, tile_dim2 = 0
        (int)uld,                                            // tensor_dim0_stride[31:0]
        0, 0                                                 // dim0_stride hi | dim1_stride
    };
    v4i z4 = {0, 0, 0, 0};
#if __clang_major__ >= 23
    v8i z8 = {0, 0, 0, 0, 0, 0, 0, 0};
    __builtin_amdgcn_tensor_load_to_lds(g0, g1, z4, z4, z8, 0);
#else
    __builtin_amdgcn_tensor_load_to_lds(g0, g1, z4, z4, 0);
#endif
}

// ---------------------------------------------------------------------------
// f32 -> bf16 pack (vectorized x4)
// ---------------------------------------------------------------------------
__global__ void __launch_bounds__(256) cvt_f32_to_bf16(const float* __restrict__ in,
                                                       unsigned short* __restrict__ out,
                                                       int n4) {
    int i = blockIdx.x * blockDim.x + threadIdx.x;
    if (i < n4) {
        v4f f = ((const v4f*)in)[i];
        v4us o;
        o.x = f2bf(f.x); o.y = f2bf(f.y); o.z = f2bf(f.z); o.w = f2bf(f.w);
        ((v4us*)out)[i] = o;
    }
}

// ---------------------------------------------------------------------------
// Pass 1: u = sigmoid(h W_u^T + b_u), rh = sigmoid(h W_r^T + b_r) * h_prev
// Block tile 128x64, 8 waves (4x2), wave tile 32x32, dual accumulators
// sharing the A (h_prev) fragments between the W_u and W_r WMMAs.
// TDM double-buffered tile staging: wave 0 issues 3 descriptors/iter.
// ---------------------------------------------------------------------------
__global__ void __launch_bounds__(256) gates_kernel(
    const unsigned short* __restrict__ hbf,    // [B,H]  bf16
    const unsigned short* __restrict__ wubf,   // [H,H]  bf16
    const unsigned short* __restrict__ wrbf,   // [H,H]  bf16
    const float* __restrict__ b_u,
    const float* __restrict__ b_r,
    const float* __restrict__ hprev,           // [B,H]  f32
    float* __restrict__ u_out,                 // [B,H]  f32 (ws)
    unsigned short* __restrict__ rh_out)       // [B,H]  bf16 (ws)
{
    __shared__ alignas(16) unsigned short sA [2][128 * LDS_STRIDE];
    __shared__ alignas(16) unsigned short sBu[2][ 64 * LDS_STRIDE];
    __shared__ alignas(16) unsigned short sBr[2][ 64 * LDS_STRIDE];

    const int tid  = threadIdx.x;
    const int wave = tid >> 5;
    const int lane = tid & 31;
    const int wm   = wave >> 1;            // 0..3 : 32-row slice
    const int wn   = wave & 1;             // 0..1 : 32-col slice
    const int rowBlock = blockIdx.x * 128;
    const int colBlock = blockIdx.y * 64;
    const int lr = lane & 15;
    const int c0 = (lane >> 4) * 8;
    const bool issuer = (wave == 0);

    auto issue = [&](int k0, int buf) {
        tdm_load_tile(&hbf [(size_t)rowBlock * HDIM + k0], lds_off_of(&sA [buf][0]), 128, HDIM);
        tdm_load_tile(&wubf[(size_t)colBlock * HDIM + k0], lds_off_of(&sBu[buf][0]),  64, HDIM);
        tdm_load_tile(&wrbf[(size_t)colBlock * HDIM + k0], lds_off_of(&sBr[buf][0]),  64, HDIM);
    };

    v8f accU[2][2] = {};
    v8f accR[2][2] = {};

    if (issuer) issue(0, 0);

    const int NIT = HDIM / 32;
    for (int it = 0; it < NIT; ++it) {
        const int cur = it & 1;
        if (issuer) {
            if (it + 1 < NIT) {
                issue((it + 1) * 32, cur ^ 1);               // prefetch next tiles
                __builtin_amdgcn_s_wait_tensorcnt(3);        // current 3 tiles landed
            } else {
                __builtin_amdgcn_s_wait_tensorcnt(0);
            }
        }
        __syncthreads();                                     // release compute waves

        v16bf a[2], bu[2], br[2];
        #pragma unroll
        for (int mf = 0; mf < 2; ++mf)
            a[mf] = load_frag(sA[cur], wm * 32 + mf * 16 + lr, c0);
        #pragma unroll
        for (int nf = 0; nf < 2; ++nf) {
            bu[nf] = load_frag(sBu[cur], wn * 32 + nf * 16 + lr, c0);
            br[nf] = load_frag(sBr[cur], wn * 32 + nf * 16 + lr, c0);
        }
        #pragma unroll
        for (int mf = 0; mf < 2; ++mf) {
            #pragma unroll
            for (int nf = 0; nf < 2; ++nf) {
                accU[mf][nf] = __builtin_amdgcn_wmma_f32_16x16x32_bf16(
                    false, a[mf], false, bu[nf], (short)0, accU[mf][nf], false, false);
                accR[mf][nf] = __builtin_amdgcn_wmma_f32_16x16x32_bf16(
                    false, a[mf], false, br[nf], (short)0, accR[mf][nf], false, false);
            }
        }
        __syncthreads();                                     // tiles consumed -> reusable
    }

    // Epilogue: C/D layout -> lane(0..15): N=lane, M=vgpr; lane(16..31): N=lane-16, M=8+vgpr
    const int col    = lane & 15;
    const int rowOff = (lane >> 4) * 8;
    #pragma unroll
    for (int nf = 0; nf < 2; ++nf) {
        int gn = colBlock + wn * 32 + nf * 16 + col;
        float bub = b_u[gn];
        float brb = b_r[gn];
        #pragma unroll
        for (int mf = 0; mf < 2; ++mf) {
            #pragma unroll
            for (int e = 0; e < 8; ++e) {
                int gm = rowBlock + wm * 32 + mf * 16 + rowOff + e;
                size_t idx = (size_t)gm * HDIM + gn;
                float uu = sigmoidf(accU[mf][nf][e] + bub);
                float rr = sigmoidf(accR[mf][nf][e] + brb);
                u_out[idx]  = uu;
                rh_out[idx] = f2bf(rr * hprev[idx]);
            }
        }
    }
}

// ---------------------------------------------------------------------------
// Pass 2 helper: accumulate A[rowBlock:+128, :K] @ B[colBlock:+128, :K]^T
// with TDM double-buffered staging (2 descriptors/iter issued by wave 0).
// ---------------------------------------------------------------------------
__device__ __forceinline__ void gemm_phase_tdm(
    const unsigned short* __restrict__ Ag, const unsigned short* __restrict__ Bg,
    int ld, int K,
    unsigned short (*sA)[128 * LDS_STRIDE], unsigned short (*sB)[128 * LDS_STRIDE],
    bool issuer, int wm, int wn, int lr, int c0,
    int rowBlock, int colBlock, v8f acc[4][2])
{
    auto issue = [&](int k0, int buf) {
        tdm_load_tile(&Ag[(size_t)rowBlock * ld + k0], lds_off_of(&sA[buf][0]), 128, ld);
        tdm_load_tile(&Bg[(size_t)colBlock * ld + k0], lds_off_of(&sB[buf][0]), 128, ld);
    };

    if (issuer) issue(0, 0);

    const int NIT = K / 32;
    for (int it = 0; it < NIT; ++it) {
        const int cur = it & 1;
        if (issuer) {
            if (it + 1 < NIT) {
                issue((it + 1) * 32, cur ^ 1);
                __builtin_amdgcn_s_wait_tensorcnt(2);
            } else {
                __builtin_amdgcn_s_wait_tensorcnt(0);
            }
        }
        __syncthreads();

        v16bf a[4], b[2];
        #pragma unroll
        for (int mf = 0; mf < 4; ++mf)
            a[mf] = load_frag(sA[cur], wm * 64 + mf * 16 + lr, c0);
        #pragma unroll
        for (int nf = 0; nf < 2; ++nf)
            b[nf] = load_frag(sB[cur], wn * 32 + nf * 16 + lr, c0);
        #pragma unroll
        for (int mf = 0; mf < 4; ++mf)
            #pragma unroll
            for (int nf = 0; nf < 2; ++nf)
                acc[mf][nf] = __builtin_amdgcn_wmma_f32_16x16x32_bf16(
                    false, a[mf], false, b[nf], (short)0, acc[mf][nf], false, false);
        __syncthreads();
    }
}

// ---------------------------------------------------------------------------
// Pass 2: h_cand = tanh(x W_hx^T + rh W_hh^T + b_h); h_next = (1-u)h + u c
// Block tile 128x128, 8 waves (2x4), wave tile 64x32.
// ---------------------------------------------------------------------------
__global__ void __launch_bounds__(256) cand_kernel(
    const unsigned short* __restrict__ xbf,    // [B,IN]  bf16
    const unsigned short* __restrict__ whxbf,  // [H,IN]  bf16
    const unsigned short* __restrict__ rhbf,   // [B,H]   bf16
    const unsigned short* __restrict__ whhbf,  // [H,H]   bf16
    const float* __restrict__ b_h,
    const float* __restrict__ u_ws,            // [B,H]   f32
    const float* __restrict__ hprev,           // [B,H]   f32
    float* __restrict__ out)                   // [2,B,H] f32
{
    __shared__ alignas(16) unsigned short sA[2][128 * LDS_STRIDE];
    __shared__ alignas(16) unsigned short sB[2][128 * LDS_STRIDE];

    const int tid  = threadIdx.x;
    const int wave = tid >> 5;
    const int lane = tid & 31;
    const int wm   = wave >> 2;            // 0..1 : 64-row slice
    const int wn   = wave & 3;             // 0..3 : 32-col slice
    const int rowBlock = blockIdx.x * 128;
    const int colBlock = blockIdx.y * 128;
    const int lr = lane & 15;
    const int c0 = (lane >> 4) * 8;
    const bool issuer = (wave == 0);

    v8f acc[4][2] = {};

    gemm_phase_tdm(xbf,  whxbf, INDIM, INDIM, sA, sB, issuer, wm, wn, lr, c0,
                   rowBlock, colBlock, acc);
    gemm_phase_tdm(rhbf, whhbf, HDIM,  HDIM,  sA, sB, issuer, wm, wn, lr, c0,
                   rowBlock, colBlock, acc);

    const int col    = lane & 15;
    const int rowOff = (lane >> 4) * 8;
    const size_t BH = (size_t)BDIM * HDIM;
    #pragma unroll
    for (int nf = 0; nf < 2; ++nf) {
        int gn = colBlock + wn * 32 + nf * 16 + col;
        float bhb = b_h[gn];
        #pragma unroll
        for (int mf = 0; mf < 4; ++mf) {
            #pragma unroll
            for (int e = 0; e < 8; ++e) {
                int gm = rowBlock + wm * 64 + mf * 16 + rowOff + e;
                size_t idx = (size_t)gm * HDIM + gn;
                float c  = tanhf(acc[mf][nf][e] + bhb);
                float u  = u_ws[idx];
                float hp = hprev[idx];
                float hn = fmaf(u, c - hp, hp);   // (1-u)*hp + u*c
                out[idx]      = hn;
                out[BH + idx] = hn;
            }
        }
    }
}

// ---------------------------------------------------------------------------
extern "C" void kernel_launch(void* const* d_in, const int* in_sizes, int n_in,
                              void* d_out, int out_size, void* d_ws, size_t ws_size,
                              hipStream_t stream) {
    const float* x     = (const float*)d_in[0];
    const float* hprev = (const float*)d_in[1];
    const float* W_u   = (const float*)d_in[2];
    const float* b_u   = (const float*)d_in[3];
    const float* W_r   = (const float*)d_in[4];
    const float* b_r   = (const float*)d_in[5];
    const float* W_hx  = (const float*)d_in[6];
    const float* W_hh  = (const float*)d_in[7];
    const float* b_h   = (const float*)d_in[8];
    float* out = (float*)d_out;

    char* ws = (char*)d_ws;
    const size_t MB = 1ull << 20;
    unsigned short* xbf   = (unsigned short*)(ws + 0);         // 16 MB  [B,IN]
    unsigned short* hbf   = (unsigned short*)(ws + 16 * MB);   // 32 MB  [B,H]
    unsigned short* wubf  = (unsigned short*)(ws + 48 * MB);   //  2 MB  [H,H]
    unsigned short* wrbf  = (unsigned short*)(ws + 50 * MB);   //  2 MB  [H,H]
    unsigned short* whxbf = (unsigned short*)(ws + 52 * MB);   //  1 MB  [H,IN]
    unsigned short* whhbf = (unsigned short*)(ws + 53 * MB);   //  2 MB  [H,H]
    float*          u_ws  = (float*)(ws + 56 * MB);            // 64 MB  [B,H]
    unsigned short* rhbf  = (unsigned short*)(ws + 120 * MB);  // 32 MB  [B,H]

    auto cvt = [&](const float* in, unsigned short* o, int n) {
        int n4 = n / 4;
        cvt_f32_to_bf16<<<(n4 + 255) / 256, 256, 0, stream>>>(in, o, n4);
    };
    cvt(x,     xbf,   BDIM * INDIM);
    cvt(hprev, hbf,   BDIM * HDIM);
    cvt(W_u,   wubf,  HDIM * HDIM);
    cvt(W_r,   wrbf,  HDIM * HDIM);
    cvt(W_hx,  whxbf, HDIM * INDIM);
    cvt(W_hh,  whhbf, HDIM * HDIM);

    dim3 g1(BDIM / 128, HDIM / 64);
    gates_kernel<<<g1, 256, 0, stream>>>(hbf, wubf, wrbf, b_u, b_r, hprev, u_ws, rhbf);

    dim3 g2(BDIM / 128, HDIM / 128);
    cand_kernel<<<g2, 256, 0, stream>>>(xbf, whxbf, rhbf, whhbf, b_h, u_ws, hprev, out);
}
